// MultiHeadAttention_3539053052118
// MI455X (gfx1250) — compile-verified, matrix operand order checked by reading, not run on hardware
//
#include <hip/hip_runtime.h>
#include <stdint.h>

// ---------------- problem constants ----------------
#define B_    2
#define S_    2048
#define HID_  2048
#define NH_   16
#define NKV_  4
#define HD_   128
#define NREP_ 4
#define M_TOT (B_*S_)        // 4096 rows for all projection GEMMs
#define NKV2_ (NKV_*HD_*2)   // 1024 kv-proj output columns

typedef unsigned short ushort_t;
typedef __bf16 bf16_t;
typedef bf16_t v16bf __attribute__((ext_vector_type(16)));
typedef float  v8f   __attribute__((ext_vector_type(8)));

union FragB16 {            // one WMMA 16-bit A/B fragment: 16 bf16 per lane
    v16bf v;
    uint4 q[2];
};

#if defined(__HIP_DEVICE_COMPILE__) && \
    __has_builtin(__builtin_amdgcn_global_load_async_to_lds_b128) && \
    __has_builtin(__builtin_amdgcn_s_wait_asynccnt)
#define HAS_ASYNC 1
#else
#define HAS_ASYNC 0
#endif

// async builtin takes pointers to 16B GCC-vector in AS1 (global) / AS3 (LDS)
typedef int v4i_t __attribute__((vector_size(16)));
typedef __attribute__((address_space(1))) v4i_t* gv4i_p;
typedef __attribute__((address_space(3))) v4i_t* lv4i_p;

// 16-byte global -> LDS copy (async on CDNA5, sync fallback otherwise)
__device__ __forceinline__ void cp16_g2l(void* lds_dst, const void* gsrc) {
#if HAS_ASYNC
    __builtin_amdgcn_global_load_async_to_lds_b128(
        (gv4i_p)gsrc, (lv4i_p)lds_dst, 0, 0);
#else
    *(uint4*)lds_dst = *(const uint4*)gsrc;
#endif
}
#if HAS_ASYNC
#define ASYNC_WAIT(n) __builtin_amdgcn_s_wait_asynccnt(n)
#else
#define ASYNC_WAIT(n)
#endif

__device__ __forceinline__ ushort_t f2bf(float f) {
    union { float f; uint32_t u; } x; x.f = f;
    uint32_t r = x.u + 0x7FFFu + ((x.u >> 16) & 1u);   // round-to-nearest-even
    return (ushort_t)(r >> 16);
}
__device__ __forceinline__ float bf2f(ushort_t h) {
    union { uint32_t u; float f; } x; x.u = ((uint32_t)h) << 16;
    return x.f;
}

// ---------------- 1) fp32 -> bf16 convert ----------------
__global__ void k_f32_to_bf16(const float* __restrict__ in,
                              ushort_t* __restrict__ out, int n) {
    int i = blockIdx.x * blockDim.x + threadIdx.x;
    int stride = gridDim.x * blockDim.x;
    for (; i < n; i += stride) out[i] = f2bf(in[i]);
}

// ---------------- 2/5) bf16 WMMA GEMM: Y[M,N] = A[M,K] @ W[N,K]^T ----------
// block = 256 threads (8 waves); tile = 128x128; K-step = 32;
// double-buffered async global->LDS staging.
#define LDA_PAD 40   // 80B row stride: 16B aligned, conflict-spreading

__global__ __launch_bounds__(256)
void k_gemm_bf16(const ushort_t* __restrict__ A, const ushort_t* __restrict__ W,
                 void* __restrict__ Y, int M, int N, int K, int outF32)
{
    __shared__ __align__(16) ushort_t sA[2][128 * LDA_PAD];
    __shared__ __align__(16) ushort_t sB[2][128 * LDA_PAD];

    const int tid  = threadIdx.x;
    const int wave = tid >> 5;
    const int lane = tid & 31;
    const int ln   = lane & 15;
    const int half = lane >> 4;
    const int m0   = blockIdx.x * 128;
    const int n0   = blockIdx.y * 128;

    // staging: 512 chunks (16B) for A tile + 512 for W tile; 4 per thread
    auto stage = [&](int buf, int k0) {
#pragma unroll
        for (int c = 0; c < 2; c++) {
            int idx = tid + c * 256;
            int row = idx >> 2;
            int col = (idx & 3) * 8;
            cp16_g2l(&sA[buf][row * LDA_PAD + col],
                     &A[(size_t)(m0 + row) * K + k0 + col]);
            cp16_g2l(&sB[buf][row * LDA_PAD + col],
                     &W[(size_t)(n0 + row) * K + k0 + col]);
        }
    };

    v8f acc[8];
#pragma unroll
    for (int j = 0; j < 8; j++)
#pragma unroll
        for (int e = 0; e < 8; e++) acc[j][e] = 0.0f;

    const int nks = K / 32;
    stage(0, 0);
    for (int ks = 0; ks < nks; ks++) {
        const int cur = ks & 1;
        if (ks + 1 < nks) {
            stage(1 - cur, (ks + 1) * 32);
            ASYNC_WAIT(4);          // previous stage's 4 ops complete (in-order)
        } else {
            ASYNC_WAIT(0);
        }
        __syncthreads();

        FragB16 fa;
        {
            int row = wave * 16 + ln;
            int cb  = half ? 8 : 0;
            fa.q[0] = *(const uint4*)&sA[cur][row * LDA_PAD + cb];
            fa.q[1] = *(const uint4*)&sA[cur][row * LDA_PAD + 16 + cb];
        }
#pragma unroll
        for (int j = 0; j < 8; j++) {
            FragB16 fb;
            int row = j * 16 + ln;
            int cb  = half ? 8 : 0;
            fb.q[0] = *(const uint4*)&sB[cur][row * LDA_PAD + cb];
            fb.q[1] = *(const uint4*)&sB[cur][row * LDA_PAD + 16 + cb];
            acc[j] = __builtin_amdgcn_wmma_f32_16x16x32_bf16(
                false, fa.v, false, fb.v, (short)0, acc[j], false, false);
        }
        __syncthreads();            // safe to overwrite buf `cur` next iter
    }

    // epilogue: C layout — VGPR r holds row (r + 8*half), col = j*16 + ln
    const int mrow = m0 + wave * 16 + (half ? 8 : 0);
#pragma unroll
    for (int j = 0; j < 8; j++) {
        int col = n0 + j * 16 + ln;
        if (outF32) {
            float* O = (float*)Y;
#pragma unroll
            for (int r = 0; r < 8; r++)
                O[(size_t)(mrow + r) * N + col] = acc[j][r];
        } else {
            ushort_t* O = (ushort_t*)Y;
#pragma unroll
            for (int r = 0; r < 8; r++)
                O[(size_t)(mrow + r) * N + col] = f2bf(acc[j][r]);
        }
    }
}

// ---------------- 3) RoPE + layout scatter ----------------
// per token: NH*64 q-pairs + NKV*64 k-pairs + NKV*128 v-copies = 1792 items
#define PER_TOK 1792

__global__ void k_rope_scatter(const ushort_t* __restrict__ qraw,
                               const ushort_t* __restrict__ kvraw,
                               const int* __restrict__ pos_ids,
                               ushort_t* __restrict__ Qr,   // [B,NH,S,HD]
                               ushort_t* __restrict__ Kr,   // [B,NKV,S,HD]
                               ushort_t* __restrict__ Vt)   // [B,NKV,HD,S]
{
    int64_t idx   = (int64_t)blockIdx.x * blockDim.x + threadIdx.x;
    int64_t total = (int64_t)B_ * S_ * PER_TOK;
    if (idx >= total) return;
    int     item = (int)(idx % PER_TOK);
    int64_t tok  = idx / PER_TOK;
    int s = (int)(tok % S_);
    int b = (int)(tok / S_);
    int pos = pos_ids[(size_t)b * S_ + s];

    if (item < NH_ * 64) {                       // ---- Q rope ----
        int h = item >> 6, d = item & 63;
        float inv = __expf(-((float)d * (1.0f / 64.0f)) * 9.210340371976184f);
        float ang = (float)pos * inv;
        float c = __cosf(ang), sn = __sinf(ang);
        const ushort_t* src = qraw + ((size_t)(b * S_ + s)) * HID_ + h * HD_;
        float x1 = bf2f(src[d]), x2 = bf2f(src[d + 64]);
        ushort_t* dst = Qr + (((size_t)(b * NH_ + h)) * S_ + s) * HD_;
        dst[d]      = f2bf(x1 * c - x2 * sn);
        dst[d + 64] = f2bf(x2 * c + x1 * sn);
    } else if (item < NH_ * 64 + NKV_ * 64) {    // ---- K rope ----
        int it = item - NH_ * 64;
        int g = it >> 6, d = it & 63;
        float inv = __expf(-((float)d * (1.0f / 64.0f)) * 9.210340371976184f);
        float ang = (float)pos * inv;
        float c = __cosf(ang), sn = __sinf(ang);
        const ushort_t* base = kvraw + ((size_t)(b * S_ + s)) * NKV2_ + g * (HD_ * 2);
        float x1 = bf2f(base[d * 2]), x2 = bf2f(base[(d + 64) * 2]);
        ushort_t* dst = Kr + (((size_t)(b * NKV_ + g)) * S_ + s) * HD_;
        dst[d]      = f2bf(x1 * c - x2 * sn);
        dst[d + 64] = f2bf(x2 * c + x1 * sn);
    } else {                                     // ---- V transpose-copy ----
        int it = item - NH_ * 64 - NKV_ * 64;
        int g = it >> 7, d = it & 127;
        Vt[(((size_t)(b * NKV_ + g)) * HD_ + d) * S_ + s] =
            kvraw[((size_t)(b * S_ + s)) * NKV2_ + g * (HD_ * 2) + d * 2 + 1];
    }
}

// ---------------- 4) causal flash attention (WMMA) ----------------
// block = 256 threads (8 waves) handles (b, h, 128-query block);
// each wave owns 16 queries; key blocks of 32; K/V staged in
// double-buffered LDS via async copies; online softmax in f32.
#define KPAD 136   // 128+8 elems: 272B row stride, conflict-free frag reads
#define VPAD 40    // 32+8  elems: 80B row stride

__global__ __launch_bounds__(256)
void k_attention(const ushort_t* __restrict__ Qr, const ushort_t* __restrict__ Kr,
                 const ushort_t* __restrict__ Vt, ushort_t* __restrict__ ctx)
{
    __shared__ __align__(16) ushort_t sK[2][32 * KPAD];    // 32 keys x 128 d
    __shared__ __align__(16) ushort_t sV[2][128 * VPAD];   // 128 d  x 32 keys
    __shared__ __align__(16) ushort_t pbuf[8][16 * 32];    // per-wave P tile

    const int tid  = threadIdx.x;
    const int wave = tid >> 5;
    const int lane = tid & 31;
    const int ln   = lane & 15;
    const int half = lane >> 4;
    const int cb   = half ? 8 : 0;

    const int qblk = blockIdx.x & 15;           // S/128 = 16
    const int h    = (blockIdx.x >> 4) & 15;
    const int b    = blockIdx.x >> 8;
    const int g    = h >> 2;                    // h / NREP
    const int qb0   = qblk * 128;
    const int qrow0 = qb0 + wave * 16;

    const ushort_t* Qb = Qr + (((size_t)(b * NH_ + h)) * S_ + qrow0) * HD_;
    const ushort_t* Kb = Kr + ((size_t)(b * NKV_ + g)) * S_ * HD_;
    const ushort_t* Vb = Vt + ((size_t)(b * NKV_ + g)) * HD_ * S_;

    // stage one 32-key block: K tile 32x128 (512 chunks) + V tile 128x32
    // (512 chunks); 4 async ops per thread
    auto stage = [&](int buf, int key0) {
#pragma unroll
        for (int c = 0; c < 2; c++) {
            int idx = tid + c * 256;
            {   // K: row = key (0..31), col over d
                int row = idx >> 4;
                int col = (idx & 15) * 8;
                cp16_g2l(&sK[buf][row * KPAD + col],
                         &Kb[(size_t)(key0 + row) * HD_ + col]);
            }
            {   // V: row = d (0..127), col over keys
                int row = idx >> 2;
                int col = (idx & 3) * 8;
                cp16_g2l(&sV[buf][row * VPAD + col],
                         &Vb[(size_t)row * S_ + key0 + col]);
            }
        }
    };

    // Q fragments for all 4 K-steps (HD=128), loaded once from global
    FragB16 qf[4];
    {
        const ushort_t* qp = Qb + (size_t)ln * HD_;
#pragma unroll
        for (int ds = 0; ds < 4; ds++) {
            qf[ds].q[0] = *(const uint4*)&qp[ds * 32 + cb];
            qf[ds].q[1] = *(const uint4*)&qp[ds * 32 + 16 + cb];
        }
    }

    float mrow[8], lrow[8];
    v8f acc[8];
#pragma unroll
    for (int r = 0; r < 8; r++) { mrow[r] = -1e30f; lrow[r] = 0.0f; }
#pragma unroll
    for (int t = 0; t < 8; t++)
#pragma unroll
        for (int e = 0; e < 8; e++) acc[t][e] = 0.0f;

    const float scale = 0.08838834764831845f;   // 1/sqrt(128)
    const int nkb = qb0 / 32 + 4;               // causal: keys < qb0+128

    stage(0, 0);
    for (int kb = 0; kb < nkb; kb++) {
        const int key0 = kb * 32;
        const int cur  = kb & 1;
        if (kb + 1 < nkb) {
            stage(1 - cur, (kb + 1) * 32);
            ASYNC_WAIT(4);
        } else {
            ASYNC_WAIT(0);
        }
        __syncthreads();

        // ---- scores: two 16x16 tiles, 4 WMMA K-steps each ----
        v8f sc[2];
#pragma unroll
        for (int nt = 0; nt < 2; nt++) {
            v8f a;
#pragma unroll
            for (int e = 0; e < 8; e++) a[e] = 0.0f;
            const ushort_t* kp = &sK[cur][(nt * 16 + ln) * KPAD];
#pragma unroll
            for (int ds = 0; ds < 4; ds++) {
                FragB16 kf;
                kf.q[0] = *(const uint4*)&kp[ds * 32 + cb];
                kf.q[1] = *(const uint4*)&kp[ds * 32 + 16 + cb];
                a = __builtin_amdgcn_wmma_f32_16x16x32_bf16(
                    false, qf[ds].v, false, kf.v, (short)0, a, false, false);
            }
            sc[nt] = a;
        }

        // ---- scale + causal mask + online softmax stats ----
        float mnew[8];
#pragma unroll
        for (int r = 0; r < 8; r++) {
            int qi  = qrow0 + r + 8 * half;
            float s0 = sc[0][r] * scale;
            float s1 = sc[1][r] * scale;
            if (key0 + ln > qi)      s0 = -1e30f;
            if (key0 + 16 + ln > qi) s1 = -1e30f;
            sc[0][r] = s0; sc[1][r] = s1;
            float rowm = fmaxf(s0, s1);
#pragma unroll
            for (int m = 1; m < 16; m <<= 1)
                rowm = fmaxf(rowm, __shfl_xor(rowm, m, 32));
            mnew[r] = fmaxf(mrow[r], rowm);
        }
#pragma unroll
        for (int r = 0; r < 8; r++) {
            float alpha = __expf(mrow[r] - mnew[r]);
            lrow[r] *= alpha;
#pragma unroll
            for (int t = 0; t < 8; t++) acc[t][r] *= alpha;
            float p0 = __expf(sc[0][r] - mnew[r]);
            float p1 = __expf(sc[1][r] - mnew[r]);
            sc[0][r] = p0; sc[1][r] = p1;
            float ps = p0 + p1;
#pragma unroll
            for (int m = 1; m < 16; m <<= 1)
                ps += __shfl_xor(ps, m, 32);
            lrow[r] += ps;
            mrow[r]  = mnew[r];
        }

        // ---- transpose P (C-layout -> A-fragment) via wave-private LDS ----
        ushort_t* pb = pbuf[wave];
#pragma unroll
        for (int r = 0; r < 8; r++) {
            int row = r + 8 * half;
            pb[row * 32 + ln]      = f2bf(sc[0][r]);
            pb[row * 32 + 16 + ln] = f2bf(sc[1][r]);
        }
        asm volatile("" ::: "memory");   // DS in-order per wave; stop reordering
        FragB16 pf;
        pf.q[0] = *(const uint4*)&pb[ln * 32 + cb];
        pf.q[1] = *(const uint4*)&pb[ln * 32 + 16 + cb];
        asm volatile("" ::: "memory");

        // ---- ctx += P(16x32) @ V(32x128): 8 WMMAs over d-tiles ----
#pragma unroll
        for (int t = 0; t < 8; t++) {
            FragB16 vf;
            const ushort_t* vp = &sV[cur][(t * 16 + ln) * VPAD];
            vf.q[0] = *(const uint4*)&vp[cb];
            vf.q[1] = *(const uint4*)&vp[16 + cb];
            acc[t] = __builtin_amdgcn_wmma_f32_16x16x32_bf16(
                false, pf.v, false, vf.v, (short)0, acc[t], false, false);
        }
        __syncthreads();                 // safe to overwrite buf `cur`
    }

    // ---- epilogue: normalize and write ctx[B,S,NH*HD] in bf16 ----
#pragma unroll
    for (int t = 0; t < 8; t++) {
#pragma unroll
        for (int r = 0; r < 8; r++) {
            int srow = qrow0 + r + 8 * half;
            size_t off = ((size_t)(b * S_) + srow) * HID_ + h * HD_ + t * 16 + ln;
            ctx[off] = f2bf(acc[t][r] / lrow[r]);
        }
    }
}

// ---------------- launcher ----------------
extern "C" void kernel_launch(void* const* d_in, const int* in_sizes, int n_in,
                              void* d_out, int out_size, void* d_ws, size_t ws_size,
                              hipStream_t stream)
{
    const float* x   = (const float*)d_in[0];
    const int*   pos = (const int*)d_in[1];
    const float* Wq  = (const float*)d_in[2];
    const float* Wkv = (const float*)d_in[3];
    const float* Wo  = (const float*)d_in[4];

    char*  ws = (char*)d_ws;
    size_t off = 0;
    auto alloc = [&](size_t bytes) {
        void* p = ws + off;
        off += (bytes + 255) & ~(size_t)255;
        return p;
    };
    ushort_t* xb    = (ushort_t*)alloc((size_t)M_TOT * HID_  * 2);
    ushort_t* Wqb   = (ushort_t*)alloc((size_t)HID_  * HID_  * 2);
    ushort_t* Wkvb  = (ushort_t*)alloc((size_t)NKV2_ * HID_  * 2);
    ushort_t* Wob   = (ushort_t*)alloc((size_t)HID_  * HID_  * 2);
    ushort_t* qraw  = (ushort_t*)alloc((size_t)M_TOT * HID_  * 2);
    ushort_t* kvraw = (ushort_t*)alloc((size_t)M_TOT * NKV2_ * 2);
    ushort_t* Qr    = (ushort_t*)alloc((size_t)B_ * NH_  * S_ * HD_ * 2);
    ushort_t* Kr    = (ushort_t*)alloc((size_t)B_ * NKV_ * S_ * HD_ * 2);
    ushort_t* Vt    = (ushort_t*)alloc((size_t)B_ * NKV_ * S_ * HD_ * 2);
    ushort_t* ctxb  = (ushort_t*)alloc((size_t)M_TOT * HID_  * 2);

    // 1) fp32 -> bf16
    k_f32_to_bf16<<<4096, 256, 0, stream>>>(x,   xb,   M_TOT * HID_);
    k_f32_to_bf16<<<4096, 256, 0, stream>>>(Wq,  Wqb,  HID_  * HID_);
    k_f32_to_bf16<<<2048, 256, 0, stream>>>(Wkv, Wkvb, NKV2_ * HID_);
    k_f32_to_bf16<<<4096, 256, 0, stream>>>(Wo,  Wob,  HID_  * HID_);

    // 2) projections
    dim3 gq(M_TOT / 128, HID_ / 128);     // 32 x 16
    dim3 gkv(M_TOT / 128, NKV2_ / 128);   // 32 x 8
    k_gemm_bf16<<<gq,  256, 0, stream>>>(xb, Wqb,  qraw,  M_TOT, HID_,  HID_, 0);
    k_gemm_bf16<<<gkv, 256, 0, stream>>>(xb, Wkvb, kvraw, M_TOT, NKV2_, HID_, 0);

    // 3) rope + scatter
    {
        long long total = (long long)B_ * S_ * PER_TOK;
        int blocks = (int)((total + 255) / 256);
        k_rope_scatter<<<blocks, 256, 0, stream>>>(qraw, kvraw, pos, Qr, Kr, Vt);
    }

    // 4) attention
    k_attention<<<B_ * NH_ * (S_ / 128), 256, 0, stream>>>(Qr, Kr, Vt, ctxb);

    // 5) output projection (f32 out)
    k_gemm_bf16<<<gq, 256, 0, stream>>>(ctxb, Wob, d_out, M_TOT, HID_, HID_, 1);
}
